// RNN_89129161326859
// MI455X (gfx1250) — matrix operand order repaired and checked by reading, not compiled, not run on hardware
//
#include <hip/hip_runtime.h>
#include <cstddef>

#define BB   64
#define TT   512
#define DD   512
#define HH   1024
#define KTOT 1536            // DD + HH
#define G4   4096            // 4*HH

typedef __attribute__((ext_vector_type(16))) __bf16 v16bf;
typedef __attribute__((ext_vector_type(8)))  float  v8f;

static __device__ __forceinline__ unsigned f2bf(float f) {
  // round-to-nearest-even fp32 -> bf16 (returned in low 16 bits)
  unsigned u = __float_as_uint(f);
  u += 0x7FFFu + ((u >> 16) & 1u);
  return u >> 16;
}

union FragU { v16bf v; uint4 q[2]; };

// ---------------- init: zero state, combine biases ----------------
__global__ __launch_bounds__(256) void init_state(
    unsigned short* __restrict__ hbf0, unsigned short* __restrict__ hbf1,
    float* __restrict__ hf, float* __restrict__ cf, float* __restrict__ bias,
    const float* __restrict__ bx, const float* __restrict__ bh) {
  int i = blockIdx.x * 256 + threadIdx.x;
  if (i < BB * HH) {
    hbf0[i] = 0; hbf1[i] = 0; hf[i] = 0.f; cf[i] = 0.f;
  }
  if (i < G4) bias[i] = bx[i] + bh[i];
}

// ---- convert x [B,T,D] fp32 -> xb [T,B,D] bf16 (t-major, per-step contiguous) ----
__global__ __launch_bounds__(256) void convert_x(
    const float* __restrict__ x, unsigned short* __restrict__ xb) {
  size_t idx = (size_t)blockIdx.x * 256 + threadIdx.x;   // < B*T*D
  int d   = (int)(idx % DD);
  size_t r = idx / DD;
  int t   = (int)(r % TT);
  int b   = (int)(r / TT);
  float v = x[((size_t)b * TT + t) * DD + d];
  xb[((size_t)t * BB + b) * DD + d] = (unsigned short)f2bf(v);
}

// ---- convert [Wx;Wh] (row-major fp32 [KTOT,4096]) -> col-major bf16 Wc[4096][KTOT] ----
__global__ __launch_bounds__(256) void convert_w(
    const float* __restrict__ Wx, const float* __restrict__ Wh,
    unsigned short* __restrict__ Wc) {
  __shared__ unsigned short tile[32][33];
  const int kT = blockIdx.x % 48;       // 1536/32 k-tiles
  const int nT = blockIdx.x / 48;       // 4096/32 n-tiles
  const int tx = threadIdx.x & 31, ty = threadIdx.x >> 5;   // ty: 0..7
#pragma unroll
  for (int i = 0; i < 4; ++i) {
    int k = kT * 32 + ty + i * 8;
    int n = nT * 32 + tx;
    float v = (k < DD) ? Wx[(size_t)k * G4 + n] : Wh[(size_t)(k - DD) * G4 + n];
    tile[ty + i * 8][tx] = (unsigned short)f2bf(v);
  }
  __syncthreads();
#pragma unroll
  for (int i = 0; i < 4; ++i) {
    int n = nT * 32 + ty + i * 8;
    int k = kT * 32 + tx;
    Wc[(size_t)n * KTOT + k] = tile[tx][ty + i * 8];
  }
}

// ---------------- one LSTM timestep, fully fused ----------------
// grid: 64 workgroups (16 hidden columns each), 256 threads (8 waves).
// wave w: mTile = w&3 (16 batches), gate pair gp = w>>2 -> gates {2gp, 2gp+1}.
__global__ __launch_bounds__(256) void lstm_step(
    const unsigned short* __restrict__ xb,    // bf16 [T][B][D]
    const int* __restrict__ lengths,
    const unsigned short* __restrict__ Wc,    // bf16 col-major [4096][1536]
    const float* __restrict__ bias,           // bx+bh [4096]
    const unsigned short* __restrict__ h_in,  // bf16 [64][1024]
    unsigned short* __restrict__ h_out,       // bf16 [64][1024]
    float* __restrict__ hf, float* __restrict__ cf,
    float* __restrict__ out_seq, int t) {
  __shared__ float gl[4 * 64 * 17];           // [gate][row(64)][col(16)] padded

  const int tid    = threadIdx.x;
  const int lane   = tid & 31;
  const int wave   = tid >> 5;
  const int mTile  = wave & 3;
  const int gp     = wave >> 2;
  const int n0     = blockIdx.x * 16;
  const int laneLo = lane & 15;
  const int sel    = lane >> 4;               // half-wave select

  const int brow = mTile * 16 + laneLo;       // batch row this lane owns in A
  const unsigned short* xrow = xb + ((size_t)t * BB + brow) * DD + sel * 16;
  const unsigned short* hrow = h_in + brow * HH + sel * 16;

  // B-fragment base pointers: lane laneLo owns weight column (gate*HH + n0 + laneLo),
  // K contiguous; half-waves split K 0..15 / 16..31 within each 32-chunk.
  const unsigned short* bp0 = Wc + (size_t)((2 * gp + 0) * HH + n0 + laneLo) * KTOT + sel * 16;
  const unsigned short* bp1 = Wc + (size_t)((2 * gp + 1) * HH + n0 + laneLo) * KTOT + sel * 16;

  v8f acc0 = {}; v8f acc1 = {};

  // ---- K over x (input projection): bf16 direct loads ----
  for (int kc = 0; kc < DD; kc += 32) {
    FragU A;
    A.q[0] = *(const uint4*)(xrow + kc);
    A.q[1] = *(const uint4*)(xrow + kc + 8);

    FragU B0, B1;
    B0.q[0] = *(const uint4*)(bp0 + kc);
    B0.q[1] = *(const uint4*)(bp0 + kc + 8);
    B1.q[0] = *(const uint4*)(bp1 + kc);
    B1.q[1] = *(const uint4*)(bp1 + kc + 8);
    __builtin_prefetch(bp0 + kc + 32, 0, 1);
    __builtin_prefetch(bp1 + kc + 32, 0, 1);

    acc0 = __builtin_amdgcn_wmma_f32_16x16x32_bf16(false, A.v, false, B0.v,
                                                   (short)0, acc0, false, false);
    acc1 = __builtin_amdgcn_wmma_f32_16x16x32_bf16(false, A.v, false, B1.v,
                                                   (short)0, acc1, false, false);
  }

  // ---- K over h (recurrent projection): bf16 direct loads ----
  for (int kc = DD; kc < KTOT; kc += 32) {
    FragU A;
    A.q[0] = *(const uint4*)(hrow + (kc - DD));
    A.q[1] = *(const uint4*)(hrow + (kc - DD) + 8);

    FragU B0, B1;
    B0.q[0] = *(const uint4*)(bp0 + kc);
    B0.q[1] = *(const uint4*)(bp0 + kc + 8);
    B1.q[0] = *(const uint4*)(bp1 + kc);
    B1.q[1] = *(const uint4*)(bp1 + kc + 8);
    __builtin_prefetch(bp0 + kc + 32, 0, 1);
    __builtin_prefetch(bp1 + kc + 32, 0, 1);

    acc0 = __builtin_amdgcn_wmma_f32_16x16x32_bf16(false, A.v, false, B0.v,
                                                   (short)0, acc0, false, false);
    acc1 = __builtin_amdgcn_wmma_f32_16x16x32_bf16(false, A.v, false, B1.v,
                                                   (short)0, acc1, false, false);
  }

  // ---- gate exchange via LDS (C layout: VGPR r -> M = r + sel*8, N = laneLo) ----
  const int mBase = mTile * 16 + sel * 8;
#pragma unroll
  for (int r = 0; r < 8; ++r) {
    gl[((2 * gp + 0) * 64 + mBase + r) * 17 + laneLo] = acc0[r];
    gl[((2 * gp + 1) * 64 + mBase + r) * 17 + laneLo] = acc1[r];
  }
  __syncthreads();

  // ---- fused elementwise LSTM update: 1024 elements, 4 per thread ----
#pragma unroll
  for (int r = 0; r < 4; ++r) {
    int e  = tid + r * 256;
    int bb = e >> 4;
    int c  = e & 15;
    int j  = n0 + c;

    float iv = gl[(0 * 64 + bb) * 17 + c] + bias[0 * HH + j];
    float fv = gl[(1 * 64 + bb) * 17 + c] + bias[1 * HH + j];
    float ov = gl[(2 * 64 + bb) * 17 + c] + bias[2 * HH + j];
    float gv = gl[(3 * 64 + bb) * 17 + c] + bias[3 * HH + j];

    iv = 1.f / (1.f + __expf(-iv));
    fv = 1.f / (1.f + __expf(-fv));
    ov = 1.f / (1.f + __expf(-ov));
    gv = tanhf(gv);

    float co = cf[bb * HH + j];
    float ho = hf[bb * HH + j];
    float cc = fv * co + iv * gv;
    float hc = ov * tanhf(cc);
    float m  = (t < lengths[bb]) ? 1.f : 0.f;
    float cn = m * cc + (1.f - m) * co;
    float hn = m * hc + (1.f - m) * ho;

    cf[bb * HH + j] = cn;
    hf[bb * HH + j] = hn;
    h_out[bb * HH + j] = (unsigned short)f2bf(hn);
    out_seq[((size_t)bb * TT + t) * HH + j] = hn;
  }
}

// ---------------- final h_t / c_t copy ----------------
__global__ __launch_bounds__(256) void finalize_state(
    const float* __restrict__ hf, const float* __restrict__ cf,
    float* __restrict__ outTail) {
  int i = blockIdx.x * 256 + threadIdx.x;
  if (i < BB * HH) {
    outTail[i] = hf[i];
    outTail[BB * HH + i] = cf[i];
  }
}

extern "C" void kernel_launch(void* const* d_in, const int* in_sizes, int n_in,
                              void* d_out, int out_size, void* d_ws, size_t ws_size,
                              hipStream_t stream) {
  const float* x       = (const float*)d_in[0];
  const int*   lengths = (const int*)d_in[1];
  const float* Wx      = (const float*)d_in[2];
  const float* bx      = (const float*)d_in[3];
  const float* Wh      = (const float*)d_in[4];
  const float* bh      = (const float*)d_in[5];
  float* out = (float*)d_out;

  // workspace carve-up (~46 MB total, 16B aligned slices)
  char* ws = (char*)d_ws;
  unsigned short* xbuf = (unsigned short*)ws; ws += (size_t)BB * TT * DD * 2; // 32 MB
  unsigned short* Wc   = (unsigned short*)ws; ws += (size_t)G4 * KTOT * 2;   // 12 MB
  unsigned short* hbf0 = (unsigned short*)ws; ws += (size_t)BB * HH * 2;
  unsigned short* hbf1 = (unsigned short*)ws; ws += (size_t)BB * HH * 2;
  float* hf   = (float*)ws; ws += (size_t)BB * HH * 4;
  float* cf   = (float*)ws; ws += (size_t)BB * HH * 4;
  float* bias = (float*)ws; ws += (size_t)G4 * 4;

  init_state<<<(BB * HH + 255) / 256, 256, 0, stream>>>(hbf0, hbf1, hf, cf, bias, bx, bh);
  convert_x<<<(int)(((size_t)BB * TT * DD) / 256), 256, 0, stream>>>(x, xbuf);
  convert_w<<<48 * 128, 256, 0, stream>>>(Wx, Wh, Wc);

  for (int t = 0; t < TT; ++t) {
    const unsigned short* hin  = (t & 1) ? hbf1 : hbf0;
    unsigned short*       hout = (t & 1) ? hbf0 : hbf1;
    lstm_step<<<HH / 16, 256, 0, stream>>>(xbuf, lengths, Wc, bias, hin, hout,
                                           hf, cf, out, t);
  }

  finalize_state<<<(BB * HH + 255) / 256, 256, 0, stream>>>(
      hf, cf, out + (size_t)BB * TT * HH);
}
